// MHA_66451734003891
// MI455X (gfx1250) — matrix operand order repaired and checked by reading, not compiled, not run on hardware
//
#include <hip/hip_runtime.h>
#include <hip/hip_bf16.h>

// MHA forward for MI455X (gfx1250, wave32). All matmuls via
// v_wmma_f32_16x16x32_bf16 (fp32 accumulate). No LDS: WMMA fragments are
// loaded directly from global memory per the CDNA5 VGPR layouts, and the
// softmax P-tile transpose is done with shfl_xor(16) inside the wave.
//
// Workspace requirement: 48 MiB.

#define EMBED 1024
#define NHEAD 16
#define HDIM  64
#define BATCH 2
#define SEQ   2048
#define MROWS (BATCH * SEQ)   // 4096

typedef __attribute__((ext_vector_type(16))) __bf16 v16bf;
typedef __attribute__((ext_vector_type(8)))  __bf16 v8bf;
typedef __attribute__((ext_vector_type(8)))  float  v8f;

static __device__ __forceinline__ v16bf frag_cat(v8bf lo, v8bf hi) {
  return __builtin_shufflevector(lo, hi, 0,1,2,3,4,5,6,7,8,9,10,11,12,13,14,15);
}

// A fragment: 16(M) x 32(K) bf16 tile, row-major, row stride ld (elements).
// Layout: lanes 0-15 -> M=lane, K = 0..15; lanes 16-31 -> M=lane-16,
// K = 16..31. Each lane: two contiguous 16-byte loads.
static __device__ __forceinline__ v16bf load_frag_a(const __bf16* p, int ld, int lane) {
  const int hlf = lane >> 4, m = lane & 15;
  const __bf16* row = p + (size_t)m * ld;
  v8bf lo = *(const v8bf*)(row + hlf * 16);
  v8bf hi = *(const v8bf*)(row + hlf * 16 + 8);
  return frag_cat(lo, hi);
}

// B fragment: 32(K) x 16(N), with memory holding Bsrc[n][k] row-major (N rows,
// contiguous K), stride ld. Contiguous K order, same as A: lanes 0-15 hold K=0..15 for N=lane,
// lanes 16-31 hold K=16..31 for N=lane-16.
static __device__ __forceinline__ v16bf load_frag_b(const __bf16* p, int ld, int lane) {
  const int hlf = lane >> 4, n = lane & 15;
  const __bf16* row = p + (size_t)n * ld + hlf * 16;
  v8bf lo = *(const v8bf*)(row);
  v8bf hi = *(const v8bf*)(row + 8);
  return frag_cat(lo, hi);
}

#define WMMA_BF16(A, B, C) \
  __builtin_amdgcn_wmma_f32_16x16x32_bf16(false, (A), false, (B), (short)0, (C), false, false)

// ---------------------------------------------------------------------------
// fp32 -> bf16 conversion
__global__ void cvt_f32_bf16(const float* __restrict__ s, __bf16* __restrict__ d, int n) {
  int i = blockIdx.x * blockDim.x + threadIdx.x;
  int stride = gridDim.x * blockDim.x;
  for (; i < n; i += stride) d[i] = (__bf16)s[i];
}

// ---------------------------------------------------------------------------
// Y = X @ W^T + bias.  X: MxK bf16 row-major. W: NxK bf16 row-major.
// mode 0: write bf16, (B,H,S,D) layout        (Q, K projections)
// mode 1: write bf16, (B,H,D,S) layout        (V projection, pre-transposed)
// mode 2: write fp32 row-major MxN            (output projection)
// Each wave computes a 16(M) x 64(N) tile.
static __device__ __forceinline__ void store_col(v8f acc, int hlf, int m0, int n,
                                                 const float* __restrict__ bias,
                                                 void* __restrict__ Y, int N, int mode) {
  float bval = bias[n];
#pragma unroll
  for (int r = 0; r < 8; ++r) {
    int m = m0 + hlf * 8 + r;
    float v = acc[r] + bval;
    if (mode == 2) {
      ((float*)Y)[(size_t)m * N + n] = v;
    } else {
      int b = m >> 11, s = m & (SEQ - 1);    // SEQ = 2048
      int h = n >> 6,  d = n & (HDIM - 1);   // HDIM = 64
      size_t idx;
      if (mode == 0) idx = (((size_t)b * NHEAD + h) * SEQ + s) * HDIM + d;
      else           idx = (((size_t)b * NHEAD + h) * HDIM + d) * SEQ + s;
      ((__bf16*)Y)[idx] = (__bf16)v;
    }
  }
}

__global__ void __launch_bounds__(256)
gemm_xwT(const __bf16* __restrict__ X, const __bf16* __restrict__ W,
         const float* __restrict__ bias, void* __restrict__ Y,
         int M, int N, int K, int mode) {
  const int lane = threadIdx.x & 31;
  const int wave = threadIdx.x >> 5;
  const int tiles_n = N >> 6;                 // N/64
  const int total = (M >> 4) * tiles_n;
  int wt = blockIdx.x * 8 + wave;
  if (wt >= total) return;
  const int m0 = (wt / tiles_n) * 16;
  const int n0 = (wt % tiles_n) * 64;

  v8f a0 = {}, a1 = {}, a2 = {}, a3 = {};
  for (int k = 0; k < K; k += 32) {
    v16bf af = load_frag_a(X + (size_t)m0 * K + k, K, lane);
    v16bf b0 = load_frag_b(W + (size_t)(n0 +  0) * K + k, K, lane);
    v16bf b1 = load_frag_b(W + (size_t)(n0 + 16) * K + k, K, lane);
    v16bf b2 = load_frag_b(W + (size_t)(n0 + 32) * K + k, K, lane);
    v16bf b3 = load_frag_b(W + (size_t)(n0 + 48) * K + k, K, lane);
    a0 = WMMA_BF16(af, b0, a0);
    a1 = WMMA_BF16(af, b1, a1);
    a2 = WMMA_BF16(af, b2, a2);
    a3 = WMMA_BF16(af, b3, a3);
  }
  const int hlf = lane >> 4, nl = lane & 15;
  store_col(a0, hlf, m0, n0 +  0 + nl, bias, Y, N, mode);
  store_col(a1, hlf, m0, n0 + 16 + nl, bias, Y, N, mode);
  store_col(a2, hlf, m0, n0 + 32 + nl, bias, Y, N, mode);
  store_col(a3, hlf, m0, n0 + 48 + nl, bias, Y, N, mode);
}

// ---------------------------------------------------------------------------
// Causal flash attention, one wave per 16-query tile of one (b,h).
// Q,K: (B,H,S,D) bf16.  Vt: (B,H,D,S) bf16.  O: (B,S,E) bf16 (heads merged).
// Scores computed transposed: St = K @ Q^T, so the C-fragment holds a fixed
// query index per lane -> per-lane online softmax with one shfl_xor(16)
// pair-reduction; out^T accumulated as Vt @ P^T.
__global__ void __launch_bounds__(256)
flash_attn(const __bf16* __restrict__ Qm, const __bf16* __restrict__ Km,
           const __bf16* __restrict__ Vt, __bf16* __restrict__ O) {
  const int lane = threadIdx.x & 31;
  const int wave = threadIdx.x >> 5;
  const int qtiles = SEQ >> 4;                 // 128
  int wt = blockIdx.x * 8 + wave;              // 0 .. B*H*qtiles-1
  if (wt >= BATCH * NHEAD * qtiles) return;
  const int bh = wt / qtiles;
  const int q0 = (wt % qtiles) * 16;

  const __bf16* Qb = Qm + (size_t)bh * SEQ * HDIM;
  const __bf16* Kb = Km + (size_t)bh * SEQ * HDIM;
  const __bf16* Vb = Vt + (size_t)bh * HDIM * SEQ;

  const int hlf = lane >> 4, li = lane & 15;
  const int iq = q0 + li;                      // this lane's query row
  const float sc = 0.125f;                     // 1/sqrt(64)

  // Q as B-operand of St = K @ Q^T (held for the whole loop)
  v16bf bq0 = load_frag_b(Qb + (size_t)q0 * HDIM,      HDIM, lane);
  v16bf bq1 = load_frag_b(Qb + (size_t)q0 * HDIM + 32, HDIM, lane);

  v8f o0 = {}, o1 = {}, o2 = {}, o3 = {};      // out^T, d-tiles 0..3
  float mrun = -3.0e38f, lrun = 0.0f;

  for (int j0 = 0; j0 <= q0 + 15; j0 += 32) {
    v8f s0 = {}, s1 = {};
    v16bf ka;
    ka = load_frag_a(Kb + (size_t)j0 * HDIM,             HDIM, lane); s0 = WMMA_BF16(ka, bq0, s0);
    ka = load_frag_a(Kb + (size_t)j0 * HDIM + 32,        HDIM, lane); s0 = WMMA_BF16(ka, bq1, s0);
    ka = load_frag_a(Kb + (size_t)(j0 + 16) * HDIM,      HDIM, lane); s1 = WMMA_BF16(ka, bq0, s1);
    ka = load_frag_a(Kb + (size_t)(j0 + 16) * HDIM + 32, HDIM, lane); s1 = WMMA_BF16(ka, bq1, s1);

    // scale + causal mask; C layout: reg r of s0 -> key j = j0 + hlf*8 + r,
    // s1 -> j = j0+16+hlf*8+r, query fixed at iq.
    float p0[8], p1[8];
    float mloc = -3.0e38f;
#pragma unroll
    for (int r = 0; r < 8; ++r) {
      int ja = j0 + hlf * 8 + r;
      int jb = ja + 16;
      float va = (ja <= iq) ? s0[r] * sc : -3.0e38f;
      float vb = (jb <= iq) ? s1[r] * sc : -3.0e38f;
      p0[r] = va; p1[r] = vb;
      mloc = fmaxf(mloc, fmaxf(va, vb));
    }
    mloc = fmaxf(mloc, __shfl_xor(mloc, 16, 32));   // pair lanes agree on max
    float mnew = fmaxf(mrun, mloc);
    float corr = __expf(mrun - mnew);
    lrun *= corr;
#pragma unroll
    for (int r = 0; r < 8; ++r) {
      p0[r] = __expf(p0[r] - mnew);
      p1[r] = __expf(p1[r] - mnew);
      lrun += p0[r] + p1[r];
    }
#pragma unroll
    for (int r = 0; r < 8; ++r) { o0[r] *= corr; o1[r] *= corr; o2[r] *= corr; o3[r] *= corr; }
    mrun = mnew;

    // Assemble P^T B-fragment (32 keys x 16 queries) from the C-layout tiles:
    // lanes 0-15 need j = j0..j0+15 (own j0..+7, partner j0+8..+15);
    // lanes 16-31 need j = j0+16..j0+31 (partner +16..+23, own +24..+31).
    v16bf bp;
#pragma unroll
    for (int e = 0; e < 8; ++e) {
      float w0 = __shfl_xor(p0[e], 16, 32);
      float w1 = __shfl_xor(p1[e], 16, 32);
      float lo = hlf ? w1 : p0[e];
      float hi = hlf ? p1[e] : w0;
      bp[e]     = (__bf16)lo;
      bp[e + 8] = (__bf16)hi;
    }

    v16bf va;
    va = load_frag_a(Vb + (size_t) 0 * SEQ + j0, SEQ, lane); o0 = WMMA_BF16(va, bp, o0);
    va = load_frag_a(Vb + (size_t)16 * SEQ + j0, SEQ, lane); o1 = WMMA_BF16(va, bp, o1);
    va = load_frag_a(Vb + (size_t)32 * SEQ + j0, SEQ, lane); o2 = WMMA_BF16(va, bp, o2);
    va = load_frag_a(Vb + (size_t)48 * SEQ + j0, SEQ, lane); o3 = WMMA_BF16(va, bp, o3);
  }

  float ltot = lrun + __shfl_xor(lrun, 16, 32);
  float inv = 1.0f / ltot;

  // out^T C layout: lane holds query iq, d = t*16 + hlf*8 + r (contiguous in r)
  const int b = bh >> 4, h = bh & (NHEAD - 1);
  __bf16* orow = O + ((size_t)(b * SEQ + iq)) * EMBED + h * HDIM + hlf * 8;
  {
    v8bf ov;
#pragma unroll
    for (int r = 0; r < 8; ++r) ov[r] = (__bf16)(o0[r] * inv);
    *(v8bf*)(orow +  0) = ov;
#pragma unroll
    for (int r = 0; r < 8; ++r) ov[r] = (__bf16)(o1[r] * inv);
    *(v8bf*)(orow + 16) = ov;
#pragma unroll
    for (int r = 0; r < 8; ++r) ov[r] = (__bf16)(o2[r] * inv);
    *(v8bf*)(orow + 32) = ov;
#pragma unroll
    for (int r = 0; r < 8; ++r) ov[r] = (__bf16)(o3[r] * inv);
    *(v8bf*)(orow + 48) = ov;
  }
}

// ---------------------------------------------------------------------------
extern "C" void kernel_launch(void* const* d_in, const int* in_sizes, int n_in,
                              void* d_out, int out_size, void* d_ws, size_t ws_size,
                              hipStream_t stream) {
  const float* x  = (const float*)d_in[0];
  const float* Wq = (const float*)d_in[1];
  const float* bq = (const float*)d_in[2];
  const float* Wk = (const float*)d_in[3];
  const float* bk = (const float*)d_in[4];
  const float* Wv = (const float*)d_in[5];
  const float* bv = (const float*)d_in[6];
  const float* Wo = (const float*)d_in[7];
  const float* bo = (const float*)d_in[8];

  char* w = (char*)d_ws;
  const size_t MB = 1024u * 1024u;
  __bf16* xb   = (__bf16*)(w);                 //  8 MiB: x as bf16
  __bf16* wqb  = (__bf16*)(w +  8 * MB);       //  2 MiB each weight
  __bf16* wkb  = (__bf16*)(w + 10 * MB);
  __bf16* wvb  = (__bf16*)(w + 12 * MB);
  __bf16* wob  = (__bf16*)(w + 14 * MB);
  __bf16* Qb   = (__bf16*)(w + 16 * MB);       //  8 MiB (B,H,S,D)
  __bf16* Kb   = (__bf16*)(w + 24 * MB);       //  8 MiB (B,H,S,D)
  __bf16* Vtb  = (__bf16*)(w + 32 * MB);       //  8 MiB (B,H,D,S)
  __bf16* attn = (__bf16*)(w + 40 * MB);       //  8 MiB (B,S,E)

  const int NX = MROWS * EMBED;                // 4194304
  const int NW = EMBED * EMBED;                // 1048576

  cvt_f32_bf16<<<1024, 256, 0, stream>>>(x,  xb,  NX);
  cvt_f32_bf16<<< 512, 256, 0, stream>>>(Wq, wqb, NW);
  cvt_f32_bf16<<< 512, 256, 0, stream>>>(Wk, wkb, NW);
  cvt_f32_bf16<<< 512, 256, 0, stream>>>(Wv, wvb, NW);
  cvt_f32_bf16<<< 512, 256, 0, stream>>>(Wo, wob, NW);

  // (M/16)*(N/64) = 256*16 = 4096 wave-tiles, 8 waves/block -> 512 blocks
  gemm_xwT<<<512, 256, 0, stream>>>(xb, wqb, bq, (void*)Qb,  MROWS, EMBED, EMBED, 0);
  gemm_xwT<<<512, 256, 0, stream>>>(xb, wkb, bk, (void*)Kb,  MROWS, EMBED, EMBED, 0);
  gemm_xwT<<<512, 256, 0, stream>>>(xb, wvb, bv, (void*)Vtb, MROWS, EMBED, EMBED, 1);

  // B*H*(S/16) = 4096 wave-tiles, 8 waves/block -> 512 blocks
  flash_attn<<<512, 256, 0, stream>>>(Qb, Kb, Vtb, attn);

  gemm_xwT<<<512, 256, 0, stream>>>(attn, wob, bo, d_out, MROWS, EMBED, EMBED, 2);
}
